// ClothesBasedAdversarialLossWithMemoryBank_9242769621721
// MI455X (gfx1250) — matrix operand order repaired and checked by reading, not compiled, not run on hardware
//
#include <hip/hip_runtime.h>
#include <math.h>

#define Bn 1024
#define Dn 512
#define Cn 50000
constexpr float kScale = 16.0f;
constexpr float kEps   = 0.1f;

typedef __attribute__((ext_vector_type(16))) __bf16 bf16x16;
typedef __attribute__((ext_vector_type(8)))  float  f32x8;

// ---- workspace layout (bytes, all 256-aligned) ----
#define OFF_SUMS   0ull
#define OFF_COUNTS 102400000ull
#define OFF_SNEG   102600192ull
#define OFF_PSUM   102604288ull
#define OFF_INORM  102608384ull
#define OFF_MNORM  103656960ull
#define ZERO_FLOATS ((long long)(OFF_INORM / 4))   // zero sums..psum contiguously

#define LDSA_STRIDE 520   // 512 + 8 halfwords: 1040B row pitch, breaks bank alignment

__device__ __forceinline__ unsigned short f32_to_bf16(float f) {
  unsigned int u = __float_as_uint(f);
  u += 0x7FFFu + ((u >> 16) & 1u);          // round-to-nearest-even
  return (unsigned short)(u >> 16);
}
__device__ __forceinline__ float bf16lo_to_f32(unsigned int u) { return __uint_as_float(u << 16); }
__device__ __forceinline__ float bf16hi_to_f32(unsigned int u) { return __uint_as_float(u & 0xFFFF0000u); }

__device__ __forceinline__ float block_reduce_sum_256(float v) {
  __shared__ float s[8];
  for (int off = 16; off > 0; off >>= 1) v += __shfl_xor(v, off, 32);
  int wid = threadIdx.x >> 5, lane = threadIdx.x & 31;
  if (lane == 0) s[wid] = v;
  __syncthreads();
  if (threadIdx.x < 8) {
    v = s[threadIdx.x];
    for (int off = 4; off > 0; off >>= 1) v += __shfl_xor(v, off, 32);
    if (threadIdx.x == 0) s[0] = v;
  }
  __syncthreads();
  float r = s[0];
  __syncthreads();
  return r;
}

// ---------------- kernel 0: zero scratch + output ----------------
__global__ void k_zero(float* __restrict__ ws, float* __restrict__ out) {
  long long n4 = ZERO_FLOATS / 4;
  float4 z = make_float4(0.f, 0.f, 0.f, 0.f);
  for (long long i = (long long)blockIdx.x * blockDim.x + threadIdx.x; i < n4;
       i += (long long)gridDim.x * blockDim.x)
    ((float4*)ws)[i] = z;
  if (blockIdx.x == 0 && threadIdx.x == 0) out[0] = 0.f;
}

// ---------------- kernel 1: segment sums (memory update) ----------------
__global__ void k_segment(const float* __restrict__ x, const int* __restrict__ tgt,
                          float* __restrict__ sums, float* __restrict__ counts) {
  int i = blockIdx.x * blockDim.x + threadIdx.x;
  if (i >= Bn * Dn) return;
  int b = i >> 9, d = i & (Dn - 1);
  int t = tgt[b];
  atomicAdd(&sums[(size_t)t * Dn + d], x[i]);
  if (d == 0) atomicAdd(&counts[t], 1.0f);
}

// ---------------- kernel 2: l2-normalize inputs -> bf16 ----------------
__global__ void k_inorm(const float* __restrict__ x, unsigned short* __restrict__ o) {
  int b = blockIdx.x, t = threadIdx.x;
  size_t base = (size_t)b * Dn;
  float v0 = x[base + t], v1 = x[base + t + 256];
  float ss = block_reduce_sum_256(v0 * v0 + v1 * v1);
  float r = 1.0f / fmaxf(sqrtf(ss), 1e-12f);
  o[base + t]       = f32_to_bf16(v0 * r);
  o[base + t + 256] = f32_to_bf16(v1 * r);
}

// ---------------- kernel 3: class memory mean + l2-normalize -> bf16 ----------------
__global__ void k_mnorm(const float* __restrict__ sums, const float* __restrict__ counts,
                        const float* __restrict__ fmem, unsigned short* __restrict__ o) {
  int c = blockIdx.x, t = threadIdx.x;
  float cnt = counts[c];
  float inv = (cnt > 0.f) ? (1.0f / cnt) : 0.f;
  size_t base = (size_t)c * Dn;
  float v0 = (cnt > 0.f) ? sums[base + t] * inv       : fmem[base + t];
  float v1 = (cnt > 0.f) ? sums[base + t + 256] * inv : fmem[base + t + 256];
  float ss = block_reduce_sum_256(v0 * v0 + v1 * v1);
  float r = 1.0f / fmaxf(sqrtf(ss), 1e-12f);
  o[base + t]       = f32_to_bf16(v0 * r);
  o[base + t + 256] = f32_to_bf16(v1 * r);
}

// ---------------- kernel 4: WMMA GEMM + fused softmax-denominator stats ----------------
// Block = 128 threads (4 waves). A tile (16 rows x 512) staged once into LDS via
// GLOBAL_LOAD_ASYNC_TO_LDS_B128 (ASYNCcnt) and shared by all waves; each wave
// register-blocks 4 N-tiles (16Mx64N strip), so one A fragment feeds 4 WMMAs.
__global__ void __launch_bounds__(128)
k_phase1(const float* __restrict__ pmask, const float* __restrict__ counts,
         const int* __restrict__ lmem,
         const unsigned short* __restrict__ inorm, const unsigned short* __restrict__ mnorm,
         float* __restrict__ sneg, float* __restrict__ psum) {
  __shared__ unsigned short s_a[16 * LDSA_STRIDE];   // ~16.6 KB
  __shared__ float s_sneg[16], s_psum[16];
  int lane = threadIdx.x & 31;
  int wid  = threadIdx.x >> 5;
  if (threadIdx.x < 16) { s_sneg[threadIdx.x] = 0.f; s_psum[threadIdx.x] = 0.f; }

  int m0 = blockIdx.y * 16;

  // ---- async-stage A tile into LDS: 1024 chunks of 16B (64 chunks/row) ----
#pragma unroll
  for (int it = 0; it < 8; ++it) {
    int chunk = threadIdx.x + it * 128;
    int row = chunk >> 6;
    int k8  = (chunk & 63) * 8;
    const unsigned short* gp = inorm + (size_t)(m0 + row) * Dn + k8;
    unsigned int loff = (unsigned int)(size_t)&s_a[row * LDSA_STRIDE + k8];
    asm volatile("global_load_async_to_lds_b128 %0, %1, off"
                 :: "v"(loff), "v"(gp) : "memory");
  }
  asm volatile("s_wait_asynccnt 0x0" ::: "memory");
  __syncthreads();

  // A fragment source (16-bit 16x32): lanes 0-15 row M=lane, K [0,8)+[16,24);
  //                                   lanes 16-31 same rows, K [8,16)+[24,32).
  int arow = (lane & 15);
  int kA   = (lane >> 4) * 8;
  // B fragment (16-bit 32x16): lanes 0-15 col N=lane, K [0,8)+[16,24); lanes 16-31 K [8,16)+[24,32).
  int kB   = (lane >> 4) * 8;

  int n0 = (blockIdx.x * 16 + wid * 4) * 16;   // 4 consecutive N-tiles per wave
  int colRaw[4];
  const unsigned short* brow[4];
#pragma unroll
  for (int j = 0; j < 4; ++j) {
    colRaw[j] = n0 + j * 16 + (lane & 15);
    int col = colRaw[j] < Cn ? colRaw[j] : (Cn - 1);
    brow[j] = mnorm + (size_t)col * Dn + kB;
  }

  f32x8 acc[4] = {};
  for (int kk = 0; kk < Dn; kk += 32) {
    union { bf16x16 v; unsigned int u[8]; } fa;
    *(uint4*)&fa.u[0] = *(const uint4*)(&s_a[arow * LDSA_STRIDE + kk + kA]);        // ds_load_b128
    *(uint4*)&fa.u[4] = *(const uint4*)(&s_a[arow * LDSA_STRIDE + kk + kA + 16]);
    if (kk + 32 < Dn) __builtin_prefetch(brow[0] + kk + 32, 0, 0);
#pragma unroll
    for (int j = 0; j < 4; ++j) {
      union { bf16x16 v; unsigned int u[8]; } fb;
      *(uint4*)&fb.u[0] = *(const uint4*)(brow[j] + kk);
      *(uint4*)&fb.u[4] = *(const uint4*)(brow[j] + kk + 16);
      acc[j] = __builtin_amdgcn_wmma_f32_16x16x32_bf16(
          /*neg_a=*/false, fa.v, /*neg_b=*/false, fb.v,
          /*c_mod=*/(short)0, acc[j], /*reuse_a=*/false, /*reuse_b=*/false);
    }
  }

  // ---- epilogue: S_neg / pos_sum partials ----
  // C/D layout: VGPR r -> row r (lanes 0-15) / row r+8 (lanes 16-31), N = lane&15.
#pragma unroll
  for (int j = 0; j < 4; ++j) {
    bool inb = colRaw[j] < Cn;
    int colc = inb ? colRaw[j] : (Cn - 1);
    float vld = (inb && (counts[colc] > 0.f || lmem[colc] != -1)) ? 1.f : 0.f;
#pragma unroll
    for (int r = 0; r < 8; ++r) {
      int rowl = r + ((lane >> 4) << 3);
      float pm = inb ? pmask[(size_t)(m0 + rowl) * Cn + colRaw[j]] : 0.f;
      float simv = acc[j][r] * kScale;
      float e = __expf(simv);
      float sn = e * (1.f - pm) * vld;     // exp(sim) * neg
      float ps = pm * vld;                 // pos
      for (int off = 1; off < 16; off <<= 1) {   // reduce the 16 lanes of each half
        sn += __shfl_xor(sn, off, 32);
        ps += __shfl_xor(ps, off, 32);
      }
      if ((lane & 15) == 0) {
        atomicAdd(&s_sneg[rowl], sn);      // ds_add_f32
        atomicAdd(&s_psum[rowl], ps);
      }
    }
  }
  __syncthreads();
  if (threadIdx.x < 16) {
    atomicAdd(&sneg[m0 + threadIdx.x], s_sneg[threadIdx.x]);
    atomicAdd(&psum[m0 + threadIdx.x], s_psum[threadIdx.x]);
  }
}

// ---------------- kernel 5: sparse loss pass ----------------
__global__ void k_loss(const float* __restrict__ pmask, const int* __restrict__ tgt,
                       const float* __restrict__ counts, const int* __restrict__ lmem,
                       const unsigned short* __restrict__ inorm,
                       const unsigned short* __restrict__ mnorm,
                       const float* __restrict__ sneg, const float* __restrict__ psum,
                       float* __restrict__ out) {
  long long idx = (long long)blockIdx.x * blockDim.x + threadIdx.x;
  if (idx >= (long long)Bn * Cn) return;
  int c = (int)(idx % Cn);
  int b = (int)(idx / Cn);
  float pm = pmask[idx];
  float ident = (tgt[b] == c) ? 1.f : 0.f;
  if (pm == 0.f && ident == 0.f) return;
  float vld = (counts[c] > 0.f || lmem[c] != -1) ? 1.f : 0.f;
  float mask = (1.f - kEps) * ident + (kEps / psum[b]) * pm * vld;
  if (mask == 0.f) return;

  const unsigned short* ap = inorm + (size_t)b * Dn;
  const unsigned short* bp = mnorm + (size_t)c * Dn;
  float acc = 0.f;
  for (int d = 0; d < Dn; d += 4) {
    unsigned int a01 = *(const unsigned int*)(ap + d);
    unsigned int a23 = *(const unsigned int*)(ap + d + 2);
    unsigned int b01 = *(const unsigned int*)(bp + d);
    unsigned int b23 = *(const unsigned int*)(bp + d + 2);
    acc += bf16lo_to_f32(a01) * bf16lo_to_f32(b01)
         + bf16hi_to_f32(a01) * bf16hi_to_f32(b01)
         + bf16lo_to_f32(a23) * bf16lo_to_f32(b23)
         + bf16hi_to_f32(a23) * bf16hi_to_f32(b23);
  }
  float sim = acc * kScale;
  float ld  = __logf(sneg[b] + __expf(sim));
  atomicAdd(out, -mask * (sim - ld) * (1.0f / Bn));
}

extern "C" void kernel_launch(void* const* d_in, const int* in_sizes, int n_in,
                              void* d_out, int out_size, void* d_ws, size_t ws_size,
                              hipStream_t stream) {
  const float* inputs  = (const float*)d_in[0];
  const float* pmask   = (const float*)d_in[1];
  const float* fmem    = (const float*)d_in[2];
  const int*   lmem    = (const int*)d_in[3];
  const int*   targets = (const int*)d_in[4];
  float* out = (float*)d_out;

  char* ws = (char*)d_ws;
  float*          sums   = (float*)(ws + OFF_SUMS);
  float*          counts = (float*)(ws + OFF_COUNTS);
  float*          sneg   = (float*)(ws + OFF_SNEG);
  float*          psum   = (float*)(ws + OFF_PSUM);
  unsigned short* inorm  = (unsigned short*)(ws + OFF_INORM);
  unsigned short* mnorm  = (unsigned short*)(ws + OFF_MNORM);

  k_zero<<<4096, 256, 0, stream>>>((float*)ws, out);
  k_segment<<<(Bn * Dn + 255) / 256, 256, 0, stream>>>(inputs, targets, sums, counts);
  k_inorm<<<Bn, 256, 0, stream>>>(inputs, inorm);
  k_mnorm<<<Cn, 256, 0, stream>>>(sums, counts, fmem, mnorm);

  // N tiles: ceil(50000/16)=3125; 16 tiles per block (4 waves x 4 tiles)
  dim3 g1((3125 + 15) / 16, Bn / 16);   // (196, 64)
  k_phase1<<<g1, 128, 0, stream>>>(pmask, counts, lmem, inorm, mnorm, sneg, psum);

  long long total = (long long)Bn * Cn;
  k_loss<<<(unsigned)((total + 255) / 256), 256, 0, stream>>>(
      pmask, targets, counts, lmem, inorm, mnorm, sneg, psum, out);
}
